// CrossAttention_16690242912719
// MI455X (gfx1250) — compile-verified, matrix operand order checked by reading, not run on hardware
//
#include <hip/hip_runtime.h>

// ---------- types (POD ext-vectors to keep unions trivial) ----------
typedef __bf16 bf16_t;
typedef __bf16        v16bf  __attribute__((ext_vector_type(16)));
typedef float         v8f    __attribute__((ext_vector_type(8)));
typedef float         f32x4  __attribute__((ext_vector_type(4)));
typedef unsigned int  u32x4  __attribute__((ext_vector_type(4)));
typedef unsigned int  u32x2  __attribute__((ext_vector_type(2)));
typedef int           i32x4  __attribute__((ext_vector_type(4)));
typedef int           i32x8  __attribute__((ext_vector_type(8)));
// async-to-LDS builtin wants pointers to 128-bit int vectors in AS1/AS3
typedef __attribute__((address_space(1))) i32x4 gbl_i32x4;
typedef __attribute__((address_space(3))) i32x4 lds_i32x4;

#define DIMC   1024
#define HEADS  16
#define HDIM   64
#define BATCH  4
#define SEQ    1024
#define SCALE  0.125f

// ---- feature probes (fall back to plain-global paths when absent) ----
#if __has_builtin(__builtin_amdgcn_global_load_async_to_lds_b128)
#define HAVE_ASYNC 1
#else
#define HAVE_ASYNC 0
#endif

#if __has_builtin(__builtin_amdgcn_tensor_load_to_lds)
#define HAVE_TDM 1
#if __has_include(<hip/amd_detail/amd_gfx1250_TDM.h>)
#define TDM_6ARG 1   // therock/clang-23 flavor
#else
#define TDM_6ARG 0   // ROCm 7.2 clang-22 flavor (5 args)
#endif
#else
#define HAVE_TDM 0
#endif

#if __has_builtin(__builtin_amdgcn_s_wait_asynccnt)
#define WAIT_ASYNC(n) __builtin_amdgcn_s_wait_asynccnt(n)
#else
#define WAIT_ASYNC(n) asm volatile("s_wait_asynccnt %0" ::"n"(n) : "memory")
#endif

#if __has_builtin(__builtin_amdgcn_s_wait_tensorcnt)
#define WAIT_TENSOR(n) __builtin_amdgcn_s_wait_tensorcnt(n)
#else
#define WAIT_TENSOR(n) asm volatile("s_wait_tensorcnt %0" ::"n"(n) : "memory")
#endif

#define COMPILER_FENCE() asm volatile("" ::: "memory")

// Build a v16bf A/B operand from two contiguous 8x bf16 (16B) chunks.
// Per ISA 16-bit A/B layout: lane group g (=lane>>4) holds K = 8g..8g+7 in
// the first 8 vector elements and K = 16+8g..23+8g in the last 8.
__device__ __forceinline__ v16bf ld16g(const bf16_t* p0, const bf16_t* p1) {
    union { u32x4 q[2]; v16bf v; } u;
    u.q[0] = *(const u32x4*)p0;
    u.q[1] = *(const u32x4*)p1;
    return u.v;
}

__device__ __forceinline__ v8f wmma_bf16(v16bf a, v16bf b, v8f c) {
    return __builtin_amdgcn_wmma_f32_16x16x32_bf16(false, a, false, b, (short)0, c,
                                                   false, false);
}

#if HAVE_ASYNC
__device__ __forceinline__ void async_b128(const bf16_t* g, bf16_t* l) {
    __builtin_amdgcn_global_load_async_to_lds_b128(
        (gbl_i32x4*)(uintptr_t)g, (lds_i32x4*)(uintptr_t)l, 0, 0);
}
#endif

#if HAVE_TDM
// Issue a TDM 2D tile load: tile (tile_d1 rows x tile_d0 elems) of 2-byte
// elements from a row-major tensor with row stride stride0, into LDS.
__device__ __forceinline__ void tdm_load_2d(unsigned lds_addr, unsigned long long ga,
                                            unsigned tile_d0, unsigned tile_d1,
                                            unsigned tdim0, unsigned tdim1,
                                            unsigned stride0) {
    u32x4 g0;
    g0.x = 1u;                                    // count=1, user descriptor
    g0.y = lds_addr;                              // D#.lds_addr
    g0.z = (unsigned)ga;                          // global_addr[31:0]
    g0.w = (unsigned)(ga >> 32) | (2u << 30);     // global_addr[56:32] | type=2
    i32x8 g1;
    g1[0] = (int)(1u << 16);                      // data_size = 2 bytes
    g1[1] = (int)((tdim0 & 0xFFFFu) << 16);       // tensor_dim0[15:0]
    g1[2] = (int)((tdim0 >> 16) | ((tdim1 & 0xFFFFu) << 16));
    g1[3] = (int)((tdim1 >> 16) | (tile_d0 << 16));  // tile_dim0
    g1[4] = (int)tile_d1;                         // tile_dim1 (tile_dim2 = 0)
    g1[5] = (int)stride0;                         // tensor_dim0_stride[31:0]
    g1[6] = 0;
    g1[7] = 0;
    i32x4 z4 = {0, 0, 0, 0};
#if TDM_6ARG
    i32x8 z8 = {0, 0, 0, 0, 0, 0, 0, 0};
    __builtin_amdgcn_tensor_load_to_lds(g0, g1, z4, z4, z8, 0);
#else
    __builtin_amdgcn_tensor_load_to_lds(g0, g1, z4, z4, 0);
#endif
}
#endif

// ---------- f32 -> bf16 conversion ----------
__global__ void k_cvt(const float* __restrict__ src, bf16_t* __restrict__ dst, int n4) {
    int i = blockIdx.x * blockDim.x + threadIdx.x;
    if (i >= n4) return;
    f32x4 f = ((const f32x4*)src)[i];
    union { bf16_t e[4]; u32x2 u; } p;
    p.e[0] = (bf16_t)f.x; p.e[1] = (bf16_t)f.y;
    p.e[2] = (bf16_t)f.z; p.e[3] = (bf16_t)f.w;
    ((u32x2*)dst)[i] = p.u;
}

// ---------- P = A @ W^T + bias  (bf16 in, bf16 out row-major, optional K^T copy) ----
// One wave per 32x32 output tile. W operand tiles staged through LDS with
// double-buffered global_load_async_to_lds_b128 when available.
__global__ void k_gemm_qk(const bf16_t* __restrict__ A, const bf16_t* __restrict__ W,
                          const float* __restrict__ bias, bf16_t* __restrict__ outRM,
                          bf16_t* __restrict__ outT, int mode) {
#if HAVE_ASYNC
    __shared__ bf16_t wstage[2][2][32][16];   // [buf][j][lane][16 halfs] = 4 KB
#endif
    const int lane = threadIdx.x;
    const int lc = lane & 15, g = lane >> 4;
    const int r0 = blockIdx.x * 32, c0 = blockIdx.y * 32;

#if HAVE_ASYNC
    auto issue_w = [&](int buf, int k) {
        #pragma unroll
        for (int j = 0; j < 2; ++j) {
            const bf16_t* wp = W + (size_t)(c0 + j * 16 + lc) * DIMC + k;
            async_b128(wp + 8 * g,      &wstage[buf][j][lane][0]);
            async_b128(wp + 16 + 8 * g, &wstage[buf][j][lane][8]);
        }
    };
    issue_w(0, 0);
#endif

    v8f acc[2][2] = {};
    #pragma unroll 1
    for (int k = 0; k < DIMC; k += 32) {
        v16bf a[2], w[2];
        #pragma unroll
        for (int i = 0; i < 2; ++i) {
            const bf16_t* ap = A + (size_t)(r0 + i * 16 + lc) * DIMC + k;
            a[i] = ld16g(ap + 8 * g, ap + 16 + 8 * g);
            __builtin_prefetch(ap + 64, 0, 0);
        }
#if HAVE_ASYNC
        const int buf = (k >> 5) & 1;
        if (k + 32 < DIMC) { issue_w(buf ^ 1, k + 32); WAIT_ASYNC(4); }
        else               { WAIT_ASYNC(0); }
        COMPILER_FENCE();
        #pragma unroll
        for (int j = 0; j < 2; ++j)
            w[j] = ld16g(&wstage[buf][j][lane][0], &wstage[buf][j][lane][8]);
#else
        #pragma unroll
        for (int j = 0; j < 2; ++j) {
            const bf16_t* wp = W + (size_t)(c0 + j * 16 + lc) * DIMC + k;
            w[j] = ld16g(wp + 8 * g, wp + 16 + 8 * g);
            __builtin_prefetch(wp + 64, 0, 0);
        }
#endif
        #pragma unroll
        for (int i = 0; i < 2; ++i)
            #pragma unroll
            for (int j = 0; j < 2; ++j)
                acc[i][j] = wmma_bf16(a[i], w[j], acc[i][j]);
    }
    // epilogue: bias, bf16 row-major store; optional packed transposed store
    #pragma unroll
    for (int i = 0; i < 2; ++i)
        #pragma unroll
        for (int j = 0; j < 2; ++j) {
            const int col = c0 + j * 16 + lc;
            const float bv = bias[col];
            union { bf16_t e[8]; u32x4 u; } pk;
            #pragma unroll
            for (int r = 0; r < 8; ++r) {
                float v = acc[i][j][r] + bv;             // C/D: vgpr r -> row r+8g
                bf16_t bh = (bf16_t)v;
                pk.e[r] = bh;
                outRM[(size_t)(r0 + i * 16 + r + 8 * g) * DIMC + col] = bh;
            }
            if (mode) {                                   // K^T: (b,h,d,L) bf16
                const int hh = col >> 6, d = col & 63;
                const int t0 = r0 + i * 16 + 8 * g;       // 8 consecutive tokens
                const int bb = t0 >> 10, l0 = t0 & 1023;
                *(u32x4*)(outT + (size_t)(((bb * HEADS + hh) << 6) + d) * SEQ + l0) = pk.u;
            }
        }
}

// ---------- attention: S = qk^T, batch-axis softmax, attn out, O = attn*V ------
// One wave per (head, 16-row n-tile). K tiles (32 rows x 64 dims, per batch)
// are streamed into LDS by the Tensor Data Mover (double-buffered) when the
// TDM builtin exists; batch-axis softmax is pure per-lane VALU since all 4
// batches sit in matching C/D accumulator layouts.
__global__ void k_attn(const bf16_t* __restrict__ qb, const bf16_t* __restrict__ kb,
                       const bf16_t* __restrict__ kTb, float* __restrict__ attn_out,
                       bf16_t* __restrict__ ob) {
#if HAVE_TDM
    __shared__ bf16_t kt_lds[2][BATCH][32 * 64];  // 2 x 4 x 4 KB = 32 KB
#endif
    __shared__ bf16_t lds[BATCH * 16 * 32];       // attn tiles (16n x 32l) bf16

    const int lane = threadIdx.x;
    const int lc = lane & 15, g = lane >> 4;
    const int h  = blockIdx.x;                // 0..15
    const int n0 = blockIdx.y * 16;           // n-tile base

    // persistent Q A-operands: 16 rows x 64 K, per batch, split K in two halves
    v16bf aq[BATCH][2];
    #pragma unroll
    for (int b = 0; b < BATCH; ++b) {
        const bf16_t* qp = qb + (size_t)((b << 10) + n0 + lc) * DIMC + (h << 6);
        aq[b][0] = ld16g(qp + 8 * g,      qp + 16 + 8 * g);
        aq[b][1] = ld16g(qp + 32 + 8 * g, qp + 48 + 8 * g);
    }

#if HAVE_TDM
    auto tdm_issue = [&](int buf, int lbase) {
        #pragma unroll
        for (int b = 0; b < BATCH; ++b) {
            unsigned lds_off = (unsigned)(uintptr_t)&kt_lds[buf][b][0];
            unsigned long long ga = (unsigned long long)(uintptr_t)
                (kb + (size_t)((b << 10) + lbase) * DIMC + (h << 6));
            tdm_load_2d(lds_off, ga, /*tile_d0=*/HDIM, /*tile_d1=*/32,
                        /*tdim0=*/DIMC, /*tdim1=*/BATCH * SEQ, /*stride0=*/DIMC);
        }
    };
    tdm_issue(0, 0);
#endif

    v8f acc[BATCH][4] = {};   // O accumulators: 16n x 64d per batch

    #pragma unroll 1
    for (int lb = 0; lb < SEQ; lb += 32) {
#if HAVE_TDM
        const int buf = (lb >> 5) & 1;
        if (lb + 32 < SEQ) { tdm_issue(buf ^ 1, lb + 32); WAIT_TENSOR(4); }
        else               { WAIT_TENSOR(0); }
        COMPILER_FENCE();
#endif
        // ---- S tiles for two 16-wide l sub-tiles, all 4 batches ----
        v8f s[2][BATCH];
        #pragma unroll
        for (int t = 0; t < 2; ++t)
            #pragma unroll
            for (int b = 0; b < BATCH; ++b) {
#if HAVE_TDM
                const bf16_t* kp = &kt_lds[buf][b][(t * 16 + lc) * HDIM];
#else
                const bf16_t* kp =
                    kb + (size_t)((b << 10) + lb + t * 16 + lc) * DIMC + (h << 6);
#endif
                v16bf blo = ld16g(kp + 8 * g,      kp + 16 + 8 * g);
                v16bf bhi = ld16g(kp + 32 + 8 * g, kp + 48 + 8 * g);
                v8f z = {};
                z = wmma_bf16(aq[b][0], blo, z);
                z = wmma_bf16(aq[b][1], bhi, z);
                s[t][b] = z;
            }

        // ---- softmax over batch axis; emit attn (f32 global) + bf16 to LDS ----
        #pragma unroll
        for (int t = 0; t < 2; ++t)
            #pragma unroll
            for (int r = 0; r < 8; ++r) {
                float xs[BATCH], es[BATCH];
                #pragma unroll
                for (int b = 0; b < BATCH; ++b) xs[b] = s[t][b][r] * SCALE;
                float m = fmaxf(fmaxf(xs[0], xs[1]), fmaxf(xs[2], xs[3]));
                float den = 0.f;
                #pragma unroll
                for (int b = 0; b < BATCH; ++b) { es[b] = __expf(xs[b] - m); den += es[b]; }
                const float inv = 1.0f / den;
                const int nl = r + 8 * g;                 // local row 0..15
                const int l  = lb + t * 16 + lc;          // absolute l
                #pragma unroll
                for (int b = 0; b < BATCH; ++b) {
                    float a = es[b] * inv;
                    attn_out[((size_t)(b * HEADS + h) * SEQ + (n0 + nl)) * SEQ + l] = a;
                    lds[b * 512 + nl * 32 + t * 16 + lc] = (bf16_t)a;
                }
            }
        __syncthreads();

        // ---- O += attn(16x32) @ V(32x64); V == K, use K^T for contiguous B ----
        #pragma unroll
        for (int b = 0; b < BATCH; ++b) {
            const bf16_t* lp = &lds[b * 512 + lc * 32];   // A row = lane%16
            v16bf ap = ld16g(lp + 8 * g, lp + 16 + 8 * g);
            #pragma unroll
            for (int j = 0; j < 4; ++j) {
                const bf16_t* vp =
                    kTb + (size_t)(((b * HEADS + h) << 6) + j * 16 + lc) * SEQ + lb;
                v16bf bv = ld16g(vp + 8 * g, vp + 16 + 8 * g);
                acc[b][j] = wmma_bf16(ap, bv, acc[b][j]);
            }
        }
        __syncthreads();
    }

    // epilogue: O (pre-projection) as bf16 row-major (B*N, DIM)
    #pragma unroll
    for (int b = 0; b < BATCH; ++b)
        #pragma unroll
        for (int j = 0; j < 4; ++j)
            #pragma unroll
            for (int r = 0; r < 8; ++r) {
                const int n = n0 + r + 8 * g;
                ob[(size_t)((b << 10) + n) * DIMC + (h << 6) + j * 16 + lc] =
                    (bf16_t)acc[b][j][r];
            }
}

// ---------- final projection: out = O @ Wproj^T + bproj (f32 out) ----------
__global__ void k_gemm_proj(const bf16_t* __restrict__ A, const bf16_t* __restrict__ W,
                            const float* __restrict__ bias, float* __restrict__ out) {
#if HAVE_ASYNC
    __shared__ bf16_t wstage[2][2][32][16];
#endif
    const int lane = threadIdx.x;
    const int lc = lane & 15, g = lane >> 4;
    const int r0 = blockIdx.x * 32, c0 = blockIdx.y * 32;

#if HAVE_ASYNC
    auto issue_w = [&](int buf, int k) {
        #pragma unroll
        for (int j = 0; j < 2; ++j) {
            const bf16_t* wp = W + (size_t)(c0 + j * 16 + lc) * DIMC + k;
            async_b128(wp + 8 * g,      &wstage[buf][j][lane][0]);
            async_b128(wp + 16 + 8 * g, &wstage[buf][j][lane][8]);
        }
    };
    issue_w(0, 0);
#endif

    v8f acc[2][2] = {};
    #pragma unroll 1
    for (int k = 0; k < DIMC; k += 32) {
        v16bf a[2], w[2];
        #pragma unroll
        for (int i = 0; i < 2; ++i) {
            const bf16_t* ap = A + (size_t)(r0 + i * 16 + lc) * DIMC + k;
            a[i] = ld16g(ap + 8 * g, ap + 16 + 8 * g);
            __builtin_prefetch(ap + 64, 0, 0);
        }
#if HAVE_ASYNC
        const int buf = (k >> 5) & 1;
        if (k + 32 < DIMC) { issue_w(buf ^ 1, k + 32); WAIT_ASYNC(4); }
        else               { WAIT_ASYNC(0); }
        COMPILER_FENCE();
        #pragma unroll
        for (int j = 0; j < 2; ++j)
            w[j] = ld16g(&wstage[buf][j][lane][0], &wstage[buf][j][lane][8]);
#else
        #pragma unroll
        for (int j = 0; j < 2; ++j) {
            const bf16_t* wp = W + (size_t)(c0 + j * 16 + lc) * DIMC + k;
            w[j] = ld16g(wp + 8 * g, wp + 16 + 8 * g);
            __builtin_prefetch(wp + 64, 0, 0);
        }
#endif
        #pragma unroll
        for (int i = 0; i < 2; ++i)
            #pragma unroll
            for (int j = 0; j < 2; ++j)
                acc[i][j] = wmma_bf16(a[i], w[j], acc[i][j]);
    }
    #pragma unroll
    for (int i = 0; i < 2; ++i)
        #pragma unroll
        for (int j = 0; j < 2; ++j) {
            const int col = c0 + j * 16 + lc;
            const float bv = bias[col];
            #pragma unroll
            for (int r = 0; r < 8; ++r)
                out[(size_t)(r0 + i * 16 + r + 8 * g) * DIMC + col] = acc[i][j][r] + bv;
        }
}

extern "C" void kernel_launch(void* const* d_in, const int* in_sizes, int n_in,
                              void* d_out, int out_size, void* d_ws, size_t ws_size,
                              hipStream_t stream) {
    (void)in_sizes; (void)n_in; (void)out_size; (void)ws_size;
    const float* x     = (const float*)d_in[0];
    const float* y     = (const float*)d_in[1];
    const float* Wqk   = (const float*)d_in[2];
    const float* bqk   = (const float*)d_in[3];
    const float* Wproj = (const float*)d_in[4];
    const float* bproj = (const float*)d_in[5];

    float* out  = (float*)d_out;                       // (4,1024,1024)
    float* attn = out + (size_t)BATCH * SEQ * DIMC;    // (4,16,1024,1024)

    const size_t M1 = (size_t)1024 * 1024;
    bf16_t* ws     = (bf16_t*)d_ws;
    bf16_t* xb     = ws;              // 4M elts
    bf16_t* yb     = xb + 4 * M1;     // 4M
    bf16_t* wqkb   = yb + 4 * M1;     // 1M
    bf16_t* wprojb = wqkb + M1;       // 1M
    bf16_t* qb     = wprojb + M1;     // 4M
    bf16_t* kb     = qb + 4 * M1;     // 4M  (row-major KV)
    bf16_t* kTb    = kb + 4 * M1;     // 4M  (KV transposed per head)
    bf16_t* ob     = kTb + 4 * M1;    // 4M  (pre-projection O)

    // f32 -> bf16 converts
    k_cvt<<<(int)(4 * M1 / 4 + 255) / 256, 256, 0, stream>>>(x, xb, (int)(4 * M1 / 4));
    k_cvt<<<(int)(4 * M1 / 4 + 255) / 256, 256, 0, stream>>>(y, yb, (int)(4 * M1 / 4));
    k_cvt<<<(int)(M1 / 4 + 255) / 256, 256, 0, stream>>>(Wqk,   wqkb,   (int)(M1 / 4));
    k_cvt<<<(int)(M1 / 4 + 255) / 256, 256, 0, stream>>>(Wproj, wprojb, (int)(M1 / 4));

    dim3 wave(32);
    // Q = x Wqk^T + b ; KV = y Wqk^T + b (also emits K^T)
    k_gemm_qk<<<dim3(128, 32), wave, 0, stream>>>(xb, wqkb, bqk, qb, nullptr, 0);
    k_gemm_qk<<<dim3(128, 32), wave, 0, stream>>>(yb, wqkb, bqk, kb, kTb, 1);
    // attention + batch-axis softmax + attn output + O accumulation
    k_attn<<<dim3(HEADS, SEQ / 16), wave, 0, stream>>>(qb, kb, kTb, attn, ob);
    // out = O Wproj^T + bproj
    k_gemm_proj<<<dim3(128, 32), wave, 0, stream>>>(ob, wprojb, bproj, out);
}
